// CrossPatchAttentionBlock_16209206575198
// MI455X (gfx1250) — matrix operand order repaired and checked by reading, not compiled, not run on hardware
//
#include <hip/hip_runtime.h>
#include <hip/hip_bf16.h>
#include <math.h>

typedef _Float16 h16;
typedef __attribute__((ext_vector_type(16))) _Float16 v16h;
typedef __attribute__((ext_vector_type(8)))  _Float16 v8h;
typedef __attribute__((ext_vector_type(8)))  float    v8f;

constexpr int BB = 8, KK = 1024, DD = 256, HH = 8, RR = 4;
constexpr int LP = 1056;      // padded rows/batch (mult of 32): 4 regs + 1024 tokens + zero-key + pad
constexpr int LVALID = 1028;  // rows >= 1028 are zeroed in xn (row 1028 is the bias-only zero-key)
constexpr int MQ = BB * LP;   // 8448 rows for sequence GEMMs
constexpr int MM = BB * KK;   // 8192 rows for MLP GEMMs

__device__ __forceinline__ v8f zero8() {
  v8f z;
#pragma unroll
  for (int i = 0; i < 8; ++i) z[i] = 0.f;
  return z;
}

__device__ __forceinline__ v16h make_frag(const h16* p0, const h16* p1) {
  v8h lo = *reinterpret_cast<const v8h*>(p0);
  v8h hi = *reinterpret_cast<const v8h*>(p1);
  v16h r;
#pragma unroll
  for (int i = 0; i < 8; ++i) { r[i] = lo[i]; r[i + 8] = hi[i]; }
  return r;
}

// A fragment (16x32 f16, MxK): lane = m + 16*hi holds K = [16hi,16hi+16)
__device__ __forceinline__ v16h load_a_frag(const h16* A, int lda, int mbase, int k0, int lane) {
  const h16* p = A + (size_t)(mbase + (lane & 15)) * lda + k0 + (lane >> 4) * 16;
  return make_frag(p, p + 8);
}

__device__ __forceinline__ v8f wmma32(v16h a, v16h b, v8f c) {
  return __builtin_amdgcn_wmma_f32_16x16x32_f16(false, a, false, b, (short)0, c, false, false);
}

// ds_swizzle xor-lane exchange with immediate pattern (no index VGPR, unlike ds_bpermute)
template <int XM>
__device__ __forceinline__ float swz(float v) {
  return __int_as_float(
      __builtin_amdgcn_ds_swizzle(__float_as_int(v), (XM << 10) | 0x1f));
}
__device__ __forceinline__ float hmax16(float v) {  // reduce within 16-lane halves
  v = fmaxf(v, swz<8>(v));
  v = fmaxf(v, swz<4>(v));
  v = fmaxf(v, swz<2>(v));
  v = fmaxf(v, swz<1>(v));
  return v;
}
__device__ __forceinline__ float hsum16(float v) {
  v += swz<8>(v); v += swz<4>(v); v += swz<2>(v); v += swz<1>(v);
  return v;
}
__device__ __forceinline__ float wsum32(float v) {  // full wave32 reduce
  v += swz<16>(v); v += swz<8>(v); v += swz<4>(v); v += swz<2>(v); v += swz<1>(v);
  return v;
}

// ---- CDNA5 async global->LDS copy (ASYNCcnt-tracked), VGLOBAL op 98 ----
__device__ __forceinline__ void async_b128(unsigned lds_off, const void* g) {
  asm volatile("global_load_async_to_lds_b128 %0, %1, off"
               :: "v"(lds_off), "v"((unsigned long long)(uintptr_t)g)
               : "memory");
}
__device__ __forceinline__ void wait_async0() {
  asm volatile("s_wait_asynccnt 0x0" ::: "memory");
}

// ---- pack fp32 weights (N,K) into B-fragment order: lane n+16hi gets 16 contiguous K ----
__global__ void pack_w_kernel(const float* __restrict__ W, h16* __restrict__ out, int N, int Kdim) {
  const int lane = threadIdx.x;
  const int tiles_k = Kdim >> 5;
  const int t = blockIdx.x / tiles_k;
  const int s = blockIdx.x % tiles_k;
  const int n = t * 16 + (lane & 15);
  const int k0 = s * 32 + (lane >> 4) * 16;
  h16* o = out + ((size_t)blockIdx.x * 32 + lane) * 16;
  const float* w = W + (size_t)n * Kdim + k0;
#pragma unroll
  for (int e = 0; e < 16; ++e) o[e] = (h16)w[e];
}

// ---- embed select + 2D RoPE + register tokens + masks; pads rows >= 1028 with zeros ----
__global__ void prepare_kernel(const float* __restrict__ x, const int* __restrict__ coords,
                               const int* __restrict__ labels, const float* __restrict__ te,
                               const float* __restrict__ ce, const float* __restrict__ regs,
                               const float* __restrict__ rope, float* __restrict__ xcat,
                               float* __restrict__ maskadd, int* __restrict__ isctx) {
  const int j = blockIdx.x, b = blockIdx.y, d = threadIdx.x;
  const size_t row = (size_t)b * LP + j;
  float val = 0.f, mk = -1e30f;
  int ic = 1;
  if (j < RR) {
    val = regs[j * DD + d] + ce[d];
    mk = 0.f;
  } else if (j < RR + KK) {
    const int kk = j - RR;
    ic = labels[b * KK + kk] > 0;
    mk = ic ? 0.f : -1e30f;
    const int cy = coords[((size_t)b * KK + kk) * 2 + 0];
    const int cx = coords[((size_t)b * KK + kk) * 2 + 1];
    const int yi = (int)fminf(fmaxf((float)cy * (1023.0f / 224.0f), 0.f), 1023.f);
    const int xi = (int)fminf(fmaxf((float)cx * (1023.0f / 224.0f), 0.f), 1023.f);
    const int base = (d < 128) ? 0 : 128;
    const int idx = base ? yi : xi;  // first half uses x-coord cache, second half y-coord
    const int qd = (d - base) >> 1;
    const int comp = d & 1;
    const float c = rope[((size_t)idx * 64 + qd) * 2 + 0];
    const float s = rope[((size_t)idx * 64 + qd) * 2 + 1];
    const float* xin = x + ((size_t)b * KK + kk) * DD;
    const int i0 = base + 2 * qd;
    const float e0 = ic ? ce[i0] : te[i0];
    const float e1 = ic ? ce[i0 + 1] : te[i0 + 1];
    const float p0 = xin[i0] + e0, p1 = xin[i0 + 1] + e1;
    val = comp ? (p0 * s + p1 * c) : (p0 * c - p1 * s);
  } else if (j == RR + KK) {
    mk = 0.f;  // zero-key row: xcat=0 so xn=0 so k=bk, v=bv; always allowed
  }
  xcat[row * DD + d] = val;
  if (d == 0) { maskadd[row] = mk; isctx[row] = ic; }
}

// ---- LayerNorm fp32 -> f16, one wave per 256-wide row; zero pad rows ----
__global__ void ln_kernel(const float* __restrict__ in, h16* __restrict__ out,
                          const float* __restrict__ g, const float* __restrict__ bvec,
                          int rowsPerBatch, int validLen) {
  const int lane = threadIdx.x & 31;
  const size_t row = (size_t)blockIdx.x * 8 + (threadIdx.x >> 5);
  const float* p = in + row * 256;
  float v[8], s = 0.f;
#pragma unroll
  for (int i = 0; i < 8; ++i) { v[i] = p[i * 32 + lane]; s += v[i]; }
  s = wsum32(s);
  const float mean = s * (1.0f / 256.0f);
  float s2 = 0.f;
#pragma unroll
  for (int i = 0; i < 8; ++i) { float d = v[i] - mean; s2 += d * d; }
  s2 = wsum32(s2);
  const float rstd = rsqrtf(s2 * (1.0f / 256.0f) + 1e-5f);
  const bool zero = ((int)(row % rowsPerBatch)) >= validLen;
#pragma unroll
  for (int i = 0; i < 8; ++i) {
    const int d = i * 32 + lane;
    const float o = (v[i] - mean) * rstd * g[d] + bvec[d];
    out[row * 256 + d] = zero ? (h16)0.f : (h16)o;
  }
}

// ---- generic WMMA GEMM: C(M,N) = A(M,K) @ W^T + bias; block 128 = 4 waves, 64x64 tile.
//      Packed-B k-slice (4KB) is async-staged into LDS once per block (double-buffered);
//      the 4 waves share it instead of re-loading identical fragments from global. ----
enum { EPI_QKV = 0, EPI_GELU = 1, EPI_W2 = 2 };

template <int EPI>
__global__ void gemm_wmma_kernel(const h16* __restrict__ A, int Kdim, int lda,
                                 const h16* __restrict__ Bp, const float* __restrict__ bias,
                                 h16* __restrict__ oq, h16* __restrict__ ok, h16* __restrict__ ovT,
                                 h16* __restrict__ of16, int ldo,
                                 float* __restrict__ of32, const float* __restrict__ resid) {
  __shared__ alignas(16) h16 bsh[2][4 * 32 * 16];  // 2 x 4KB double buffer
  const int lane = threadIdx.x & 31;
  const int wave = threadIdx.x >> 5;
  const int mbase = (blockIdx.x * 4 + wave) * 16;
  const int nbase = blockIdx.y * 64;
  const int ktiles = Kdim >> 5;

  // each wave async-copies the fragment slice of its own n-tile (j == wave)
  const size_t twave = (size_t)((nbase >> 4) + wave);
  const unsigned dbase = (unsigned)(uintptr_t)&bsh[0][(wave * 32 + lane) * 16];
  const unsigned dstride = (unsigned)((uintptr_t)&bsh[1][0] - (uintptr_t)&bsh[0][0]);
  auto issueB = [&](int s, int buf) {
    const h16* src = Bp + ((twave * ktiles + s) * 32 + lane) * 16;
    const unsigned dst = dbase + buf * dstride;
    async_b128(dst, src);
    async_b128(dst + 16, src + 8);
  };

  v8f acc[4];
#pragma unroll
  for (int j = 0; j < 4; ++j) acc[j] = zero8();

  issueB(0, 0);
  int buf = 0;
  for (int s = 0; s < ktiles; ++s) {
    const v16h a = load_a_frag(A, lda, mbase, s * 32, lane);
    if (s + 2 < ktiles)
      __builtin_prefetch(A + (size_t)(mbase + (lane & 15)) * lda + (s + 2) * 32, 0, 1);
    wait_async0();     // my slice of current buffer has landed
    __syncthreads();   // everyone's slice landed; previous step's DS reads drained
    if (s + 1 < ktiles) issueB(s + 1, buf ^ 1);  // overlaps with WMMAs below
#pragma unroll
    for (int j = 0; j < 4; ++j) {
      const h16* bp = &bsh[buf][(j * 32 + lane) * 16];
      acc[j] = wmma32(a, make_frag(bp, bp + 8), acc[j]);
    }
    buf ^= 1;
  }

  const int hi8 = (lane >> 4) * 8, nl = lane & 15;
#pragma unroll
  for (int j = 0; j < 4; ++j) {
    const int col = nbase + j * 16 + nl;
    const float bv = bias[col];
#pragma unroll
    for (int r = 0; r < 8; ++r) {
      const size_t row = (size_t)mbase + r + hi8;
      const float v = acc[j][r] + bv;
      if (EPI == EPI_QKV) {
        if (col < 256) {
          oq[row * 256 + col] = (h16)v;
        } else if (col < 512) {
          ok[row * 256 + (col - 256)] = (h16)v;
        } else {  // v stored transposed per head: vT[b][h][hd][seq]
          const int c = col - 512;
          const size_t bidx = row / LP, jr = row % LP;
          ovT[((bidx * HH + (c >> 5)) * 32 + (c & 31)) * (size_t)LP + jr] = (h16)v;
        }
      } else if (EPI == EPI_GELU) {
        const float gl = 0.5f * v * (1.0f + erff(v * 0.70710678118654752f));
        of16[row * (size_t)ldo + col] = (h16)gl;
      } else {
        of32[row * 256 + col] = resid[row * 256 + col] + v;
      }
    }
  }
}

// ---- flash attention per (b, h, 16-query tile); hd=32 => one A-frag per q tile ----
__global__ void attn_kernel(const h16* __restrict__ q, const h16* __restrict__ k,
                            const h16* __restrict__ vT, const float* __restrict__ maskadd,
                            h16* __restrict__ out) {
  __shared__ alignas(16) h16 pbuf[2][16][32];
  const int lane = threadIdx.x & 31;
  const int wave = threadIdx.x >> 5;
  const int qt = blockIdx.x * 2 + wave;
  const int hh = blockIdx.y, bb = blockIdx.z;
  const int nl = lane & 15, hi = lane >> 4;
  const size_t rb = (size_t)bb * LP;
  const v16h qa = load_a_frag(q, 256, bb * LP + qt * 16, hh * 32, lane);
  float mstate[8], lstate[8];
#pragma unroll
  for (int r = 0; r < 8; ++r) { mstate[r] = -1e30f; lstate[r] = 0.f; }
  v8f o0 = zero8(), o1 = zero8();
  const float rs = 0.1767766952966369f;  // 1/sqrt(32)
  const h16* vbase = vT + ((size_t)bb * HH + hh) * 32 * (size_t)LP;
  for (int j0 = 0; j0 < LP; j0 += 32) {
    // scores: B-frag = k rows (contiguous 32B per lane)
    const h16* kp0 = k + (rb + j0 + nl) * 256 + hh * 32 + hi * 16;
    const h16* kp1 = kp0 + (size_t)16 * 256;
    if (j0 + 32 < LP) {
      __builtin_prefetch(kp0 + (size_t)32 * 256, 0, 1);
      __builtin_prefetch(vbase + (size_t)nl * LP + j0 + 32 + hi * 16, 0, 1);
    }
    v8f s0 = wmma32(qa, make_frag(kp0, kp0 + 8), zero8());
    v8f s1 = wmma32(qa, make_frag(kp1, kp1 + 8), zero8());
    const float mk0 = maskadd[bb * LP + j0 + nl];
    const float mk1 = maskadd[bb * LP + j0 + 16 + nl];
#pragma unroll
    for (int r = 0; r < 8; ++r) {
      const float x0 = s0[r] * rs + mk0;
      const float x1 = s1[r] * rs + mk1;
      const float mx = hmax16(fmaxf(x0, x1));
      const float mnew = fmaxf(mstate[r], mx);
      const float c = __expf(mstate[r] - mnew);
      const float p0 = __expf(x0 - mnew);
      const float p1 = __expf(x1 - mnew);
      lstate[r] = lstate[r] * c + hsum16(p0 + p1);
      mstate[r] = mnew;
      const int m = r + hi * 8;
      pbuf[wave][m][nl] = (h16)p0;
      pbuf[wave][m][16 + nl] = (h16)p1;
      o0[r] *= c;
      o1[r] *= c;
    }
    __syncthreads();  // LDS transpose of P (C-layout -> A-frag layout)
    const h16* pp = &pbuf[wave][nl][hi * 16];
    const v16h pa = make_frag(pp, pp + 8);
    const h16* vp0 = vbase + (size_t)nl * LP + j0 + hi * 16;
    const h16* vp1 = vbase + (size_t)(16 + nl) * LP + j0 + hi * 16;
    o0 = wmma32(pa, make_frag(vp0, vp0 + 8), o0);
    o1 = wmma32(pa, make_frag(vp1, vp1 + 8), o1);
    __syncthreads();
  }
#pragma unroll
  for (int r = 0; r < 8; ++r) {
    const float inv = 1.0f / lstate[r];
    const size_t row = rb + qt * 16 + r + hi * 8;
    out[row * 256 + hh * 32 + nl] = (h16)(o0[r] * inv);
    out[row * 256 + hh * 32 + 16 + nl] = (h16)(o1[r] * inv);
  }
}

// ---- dual Wo GEMM + per-row ctx/tgt select + residual (drops register/pad rows) ----
__global__ void wo_select_kernel(const h16* __restrict__ Ac, const h16* __restrict__ At,
                                 const h16* __restrict__ Bc, const h16* __restrict__ Bt,
                                 const float* __restrict__ bic, const float* __restrict__ bit,
                                 const int* __restrict__ isctx, const float* __restrict__ xcat,
                                 float* __restrict__ x2) {
  const int lane = threadIdx.x & 31;
  const int wave = threadIdx.x >> 5;
  const int mbase = (blockIdx.x * 4 + wave) * 16;
  const int nbase = blockIdx.y * 64;
  v8f ac[4], at[4];
#pragma unroll
  for (int j = 0; j < 4; ++j) { ac[j] = zero8(); at[j] = zero8(); }
  for (int s = 0; s < 8; ++s) {
    const v16h a0 = load_a_frag(Ac, 256, mbase, s * 32, lane);
    const v16h a1 = load_a_frag(At, 256, mbase, s * 32, lane);
#pragma unroll
    for (int j = 0; j < 4; ++j) {
      const size_t t = (size_t)((nbase >> 4) + j);
      const h16* b0 = Bc + ((t * 8 + s) * 32 + lane) * 16;
      const h16* b1 = Bt + ((t * 8 + s) * 32 + lane) * 16;
      ac[j] = wmma32(a0, make_frag(b0, b0 + 8), ac[j]);
      at[j] = wmma32(a1, make_frag(b1, b1 + 8), at[j]);
    }
  }
  const int hi8 = (lane >> 4) * 8, nl = lane & 15;
#pragma unroll
  for (int j = 0; j < 4; ++j) {
    const int col = nbase + j * 16 + nl;
    const float bc = bic[col], bt = bit[col];
#pragma unroll
    for (int r = 0; r < 8; ++r) {
      const size_t row = (size_t)mbase + r + hi8;
      const int jr = (int)(row % LP);
      if (jr >= RR && jr < RR + KK) {
        const float v = isctx[row] ? (ac[j][r] + bc) : (at[j][r] + bt);
        const size_t bidx = row / LP;
        x2[(bidx * KK + (jr - RR)) * 256 + col] = xcat[row * 256 + col] + v;
      }
    }
  }
}

extern "C" void kernel_launch(void* const* d_in, const int* in_sizes, int n_in,
                              void* d_out, int out_size, void* d_ws, size_t ws_size,
                              hipStream_t stream) {
  const float* x       = (const float*)d_in[0];
  const int*   coords  = (const int*)d_in[1];
  const int*   labels  = (const int*)d_in[2];
  const float* te      = (const float*)d_in[3];
  const float* ce      = (const float*)d_in[4];
  const float* regs    = (const float*)d_in[5];
  const float* rope    = (const float*)d_in[6];
  const float* norm_g  = (const float*)d_in[7];
  const float* norm_b  = (const float*)d_in[8];
  const float* cWqkv   = (const float*)d_in[9];
  const float* cbqkv   = (const float*)d_in[10];
  const float* cWo     = (const float*)d_in[11];
  const float* cbo     = (const float*)d_in[12];
  const float* tWqkv   = (const float*)d_in[13];
  const float* tbqkv   = (const float*)d_in[14];
  const float* tWo     = (const float*)d_in[15];
  const float* tbo     = (const float*)d_in[16];
  const float* mlpn_g  = (const float*)d_in[17];
  const float* mlpn_b  = (const float*)d_in[18];
  const float* W1      = (const float*)d_in[19];
  const float* b1      = (const float*)d_in[20];
  const float* W2      = (const float*)d_in[21];
  const float* b2      = (const float*)d_in[22];

  char* p = (char*)d_ws;
  size_t off = 0;
  auto A = [&](size_t n) { char* r = p + off; off += (n + 255) & ~(size_t)255; return r; };
  float* XCAT = (float*)A((size_t)MQ * 256 * 4);
  h16*   XN   = (h16*)  A((size_t)MQ * 256 * 2);
  h16*   QC   = (h16*)  A((size_t)MQ * 256 * 2);
  h16*   KC   = (h16*)  A((size_t)MQ * 256 * 2);
  h16*   VC   = (h16*)  A((size_t)MQ * 256 * 2);
  h16*   QT   = (h16*)  A((size_t)MQ * 256 * 2);
  h16*   KT   = (h16*)  A((size_t)MQ * 256 * 2);
  h16*   VTb  = (h16*)  A((size_t)MQ * 256 * 2);
  h16*   AC   = (h16*)  A((size_t)MQ * 256 * 2);
  h16*   AT   = (h16*)  A((size_t)MQ * 256 * 2);
  float* X2   = (float*)A((size_t)MM * 256 * 4);
  float* MASK = (float*)A((size_t)BB * LP * 4);
  int*   ICX  = (int*)  A((size_t)BB * LP * 4);
  h16*   PWQC = (h16*)A(768 * 256 * 2);
  h16*   PWQT = (h16*)A(768 * 256 * 2);
  h16*   PWOC = (h16*)A(256 * 256 * 2);
  h16*   PWOT = (h16*)A(256 * 256 * 2);
  h16*   PW1  = (h16*)A(1024 * 256 * 2);
  h16*   PW2  = (h16*)A(256 * 1024 * 2);
  h16*   Hb = XN;  // reuse: xn dead after QKV GEMMs
  h16*   H1 = QC;  // reuse: q/k/v region (17.3MB) dead after attention; H1 needs 16.8MB

  // pack weights to WMMA B-fragment order
  pack_w_kernel<<<384, 32, 0, stream>>>(cWqkv, PWQC, 768, 256);
  pack_w_kernel<<<384, 32, 0, stream>>>(tWqkv, PWQT, 768, 256);
  pack_w_kernel<<<128, 32, 0, stream>>>(cWo, PWOC, 256, 256);
  pack_w_kernel<<<128, 32, 0, stream>>>(tWo, PWOT, 256, 256);
  pack_w_kernel<<<512, 32, 0, stream>>>(W1, PW1, 1024, 256);
  pack_w_kernel<<<512, 32, 0, stream>>>(W2, PW2, 256, 1024);

  prepare_kernel<<<dim3(LP, BB), 256, 0, stream>>>(x, coords, labels, te, ce, regs, rope,
                                                   XCAT, MASK, ICX);
  ln_kernel<<<MQ / 8, 256, 0, stream>>>(XCAT, XN, norm_g, norm_b, LP, LVALID);

  gemm_wmma_kernel<EPI_QKV><<<dim3(MQ / 64, 12), 128, 0, stream>>>(
      XN, 256, 256, PWQC, cbqkv, QC, KC, VC, nullptr, 0, nullptr, nullptr);
  gemm_wmma_kernel<EPI_QKV><<<dim3(MQ / 64, 12), 128, 0, stream>>>(
      XN, 256, 256, PWQT, tbqkv, QT, KT, VTb, nullptr, 0, nullptr, nullptr);

  attn_kernel<<<dim3(LP / 32, HH, BB), 64, 0, stream>>>(QC, KC, VC, MASK, AC);
  attn_kernel<<<dim3(LP / 32, HH, BB), 64, 0, stream>>>(QT, KT, VTb, MASK, AT);

  wo_select_kernel<<<dim3(MQ / 64, 4), 128, 0, stream>>>(AC, AT, PWOC, PWOT, cbo, tbo,
                                                         ICX, XCAT, X2);

  ln_kernel<<<MM / 8, 256, 0, stream>>>(X2, Hb, mlpn_g, mlpn_b, KK, KK);
  gemm_wmma_kernel<EPI_GELU><<<dim3(MM / 64, 16), 128, 0, stream>>>(
      Hb, 256, 256, PW1, b1, nullptr, nullptr, nullptr, H1, 1024, nullptr, nullptr);
  gemm_wmma_kernel<EPI_W2><<<dim3(MM / 64, 4), 128, 0, stream>>>(
      H1, 1024, 1024, PW2, b2, nullptr, nullptr, nullptr, nullptr, 0, (float*)d_out, X2);
}
